// CoSwinAttnBlock_76785425318143
// MI455X (gfx1250) — matrix operand order repaired and checked by reading, not compiled, not run on hardware
//
#include <hip/hip_runtime.h>
#include <hip/hip_bf16.h>
#include <math.h>

// ---------------- CDNA5 WMMA types ----------------
typedef __attribute__((ext_vector_type(16))) _Float16 v16h;
typedef __attribute__((ext_vector_type(8)))  _Float16 v8h;
typedef __attribute__((ext_vector_type(8)))  float    v8f;

union H16 { v16h v; _Float16 h[16]; };

// Problem constants (from reference)
#define NB    4
#define HH    128
#define WWI   256
#define CC    96
#define NHEAD 6
#define HD    16
#define NWPI  512          // windows per image (16*32)
#define NWIN  2048         // windows per direction (B * 512)
#define TOK   64           // tokens per window
#define HID   384

static __device__ inline v8f zero8() {
  v8f z; 
#pragma unroll
  for (int i = 0; i < 8; ++i) z[i] = 0.f;
  return z;
}

static __device__ inline v8f wmma_f16(v16h a, v16h b, v8f c) {
  return __builtin_amdgcn_wmma_f32_16x16x32_f16(false, a, false, b, (short)0, c, false, false);
}

// A-matrix fragment (16x32 f16), row-major source with leading dim ld.
// lane L<16 holds row m=L, K in {0..15}; lane L>=16 same row group, K {16..31}.
static __device__ inline v16h ldA(const _Float16* M, int ld, int r0, int k0, int lane) {
  int m = lane & 15, hi = lane >> 4;
  const _Float16* p = M + (size_t)(r0 + m) * ld + k0 + hi * 16;
  H16 A;
  *(v8h*)&A.h[0] = *(const v8h*)p;
  *(v8h*)&A.h[8] = *(const v8h*)(p + 8);
  return A.v;
}

// B-matrix fragment (32x16 f16) where B[k][n] = W[n0+n][k0+k] (W row-major, ld).
// lane L: n = L&15; K rows (L>=16?16:0)+i -> 16 contiguous halves per lane.
static __device__ inline v16h ldB_rowk(const _Float16* W, int ld, int n0, int k0, int lane) {
  int n = n0 + (lane & 15), hi = lane >> 4;
  const _Float16* p = W + (size_t)n * ld + k0 + hi * 16;
  H16 B;
  *(v8h*)&B.h[0] = *(const v8h*)p;
  *(v8h*)&B.h[8] = *(const v8h*)(p + 8);
  return B.v;
}

// A fragment for head-sliced Q (K=16 real, padded to 32 with zeros).
static __device__ inline v16h ldA_headQ(const _Float16* Q, int r0, int h, int lane) {
  int m = lane & 15, hi = lane >> 4;
  const _Float16* p = Q + (size_t)(r0 + m) * CC + h * HD;
  H16 A;
  if (hi == 0) {
    *(v8h*)&A.h[0] = *(const v8h*)p;
    *(v8h*)&A.h[8] = *(const v8h*)(p + 8);
  } else {
#pragma unroll
  for (int i = 0; i < 16; ++i) A.h[i] = (_Float16)0.f;
  }
  return A.v;
}

// B fragment for K^T head slice: B[k][n] = Klds[n0+n][h*16+k], K rows 16..31 zero.
static __device__ inline v16h ldB_headK(const _Float16* K, int n0, int h, int lane) {
  H16 B;
  if (lane < 16) {
    const _Float16* p = K + (size_t)(n0 + lane) * CC + h * HD;
    *(v8h*)&B.h[0] = *(const v8h*)p;
    *(v8h*)&B.h[8] = *(const v8h*)(p + 8);
  } else {
#pragma unroll
    for (int i = 0; i < 16; ++i) B.h[i] = (_Float16)0.f;
  }
  return B.v;
}

// ---------------- kernel 1: shift-window mask (512,64,64) ----------------
static __device__ inline int regid(int v, int e1, int e2) { return v < e1 ? 0 : (v < e2 ? 1 : 2); }

__global__ void ck_mask(float* __restrict__ mask) {
  int idx = blockIdx.x * blockDim.x + threadIdx.x;
  if (idx >= NWPI * TOK * TOK) return;
  int k = idx & 63, q = (idx >> 6) & 63, wi = idx >> 12;
  int wy = wi >> 5, wx = wi & 31;
  int qy = wy * 8 + (q >> 3), qx = wx * 8 + (q & 7);
  int ky = wy * 8 + (k >> 3), kx = wx * 8 + (k & 7);
  int idq = regid(qy, 120, 124) * 3 + regid(qx, 248, 252);
  int idk = regid(ky, 120, 124) * 3 + regid(kx, 248, 252);
  mask[idx] = (idq != idk) ? -100.f : 0.f;
}

// ---------------- kernel 2: weight f32 -> f16 ----------------
__global__ void ck_f2h(const float* __restrict__ s, _Float16* __restrict__ d, int n) {
  int i = blockIdx.x * blockDim.x + threadIdx.x;
  if (i < n) d[i] = (_Float16)s[i];
}

// ---------------- kernel 3: LN1 + disparity warp + roll + window partition ----------------
static __device__ inline void ln_store(const float* __restrict__ row,
                                       const float* __restrict__ g,
                                       const float* __restrict__ bt,
                                       _Float16* __restrict__ out, int lane) {
  float v0 = row[lane], v1 = row[lane + 32], v2 = row[lane + 64];
  float s = v0 + v1 + v2;
  float s2 = v0 * v0 + v1 * v1 + v2 * v2;
#pragma unroll
  for (int o = 16; o > 0; o >>= 1) {
    s  += __shfl_xor(s, o, 32);
    s2 += __shfl_xor(s2, o, 32);
  }
  float m   = s * (1.f / 96.f);
  float var = s2 * (1.f / 96.f) - m * m;
  float inv = rsqrtf(var + 1e-5f);
  out[lane]      = (_Float16)((v0 - m) * inv * g[lane]      + bt[lane]);
  out[lane + 32] = (_Float16)((v1 - m) * inv * g[lane + 32] + bt[lane + 32]);
  out[lane + 64] = (_Float16)((v2 - m) * inv * g[lane + 64] + bt[lane + 64]);
}

__global__ __launch_bounds__(128)
void ck_prep(const float* __restrict__ xl, const float* __restrict__ xr,
             const int* __restrict__ dl, const int* __restrict__ dr,
             const float* __restrict__ g, const float* __restrict__ bt,
             _Float16* __restrict__ Xlw, _Float16* __restrict__ Xrw,
             _Float16* __restrict__ Xlsw, _Float16* __restrict__ Xrsw) {
  int gw   = blockIdx.x * 4 + (threadIdx.x >> 5);   // one wave per window token
  int lane = threadIdx.x & 31;
  int dir  = gw >> 17;                 // 2048*64 tokens per direction
  int rem  = gw & 131071;
  int widx = rem >> 6, t = rem & 63;
  int b  = widx >> 9, wi = widx & 511;
  int wy = wi >> 5,  wx = wi & 31;
  int y = ((wy << 3) + (t >> 3) + 4) & 127;        // roll(-ss) inverse source
  int x = ((wx << 3) + (t & 7) + 4) & 255;

  const float* X = dir ? xr : xl;
  const int*   D = dir ? dl : dr;                  // left uses d_right, right uses d_left
  int d  = D[(b << 15) + (y << 8) + x];
  int xs = dir ? (x - d) : (x + d);
  xs = xs < 0 ? 0 : (xs > 255 ? 255 : xs);

  _Float16* Wbuf = dir ? Xrw  : Xlw;
  _Float16* Sbuf = dir ? Xrsw : Xlsw;
  size_t dsto = ((size_t)widx * TOK + t) * CC;

  ln_store(X + (size_t)((b << 15) + (y << 8) + x)  * CC, g, bt, Wbuf + dsto, lane);
  ln_store(X + (size_t)((b << 15) + (y << 8) + xs) * CC, g, bt, Sbuf + dsto, lane);
}

// ---------------- kernel 4: per-window fused attention + fuse GEMM ----------------
// smem layout (bytes)
#define SM_XW   0
#define SM_XS   12288
#define SM_Q    24576
#define SM_K    36864
#define SM_VT   49152
#define SM_P    61440
#define SM_O    69632
#define SM_Y    81920
#define SM_S    94208
#define SM_ATTN 110592

__global__ __launch_bounds__(128)
void ck_attn(const _Float16* __restrict__ Xlw, const _Float16* __restrict__ Xrw,
             const _Float16* __restrict__ Xlsw, const _Float16* __restrict__ Xrsw,
             const float* __restrict__ mask,
             const _Float16* __restrict__ qw, const float* __restrict__ qb,
             const _Float16* __restrict__ kvw, const float* __restrict__ kvb,
             const _Float16* __restrict__ pw, const float* __restrict__ pb,
             const _Float16* __restrict__ fw, const float* __restrict__ fb,
             float* __restrict__ Fl, float* __restrict__ Fr) {
  extern __shared__ char smem[];
  _Float16* sXw = (_Float16*)(smem + SM_XW);
  _Float16* sXs = (_Float16*)(smem + SM_XS);
  _Float16* sQ  = (_Float16*)(smem + SM_Q);
  _Float16* sK  = (_Float16*)(smem + SM_K);
  _Float16* sVt = (_Float16*)(smem + SM_VT);   // V transposed: [96][64]
  _Float16* sP  = (_Float16*)(smem + SM_P);
  _Float16* sO  = (_Float16*)(smem + SM_O);
  _Float16* sY  = (_Float16*)(smem + SM_Y);
  float*    sS  = (float*)   (smem + SM_S);

  int dir  = blockIdx.x >> 11;
  int widx = blockIdx.x & 2047;
  const _Float16* gXw = (dir ? Xrw  : Xlw)  + (size_t)widx * TOK * CC;
  const _Float16* gXs = (dir ? Xlsw : Xrsw) + (size_t)widx * TOK * CC;  // cross view
  float*          gF  = (dir ? Fr   : Fl)   + (size_t)widx * TOK * CC;

  int tid = threadIdx.x, wave = tid >> 5, lane = tid & 31;
  int r0 = wave * 16;
  int nlo = lane & 15, hi8 = (lane >> 4) << 3;

  // stage activations into LDS
  for (int i = tid; i < TOK * CC / 8; i += 128) {
    ((v8h*)sXw)[i] = ((const v8h*)gXw)[i];
    ((v8h*)sXs)[i] = ((const v8h*)gXs)[i];
  }
  __syncthreads();

  // Q = Xw @ qw^T + qb, scaled by hd^-0.5
  for (int ct = 0; ct < 6; ++ct) {
    v8f acc = zero8();
#pragma unroll
    for (int kk = 0; kk < 3; ++kk)
      acc = wmma_f16(ldA(sXw, CC, r0, kk * 32, lane),
                     ldB_rowk(qw, CC, ct * 16, kk * 32, lane), acc);
    int n = ct * 16 + nlo; float bb = qb[n]; int rb = r0 + hi8;
#pragma unroll
    for (int r = 0; r < 8; ++r) sQ[(rb + r) * CC + n] = (_Float16)((acc[r] + bb) * 0.25f);
  }

  // KV = Xsel @ kvw^T + kvb ; cols 0..95 -> K, 96..191 -> V (transposed store)
  for (int ct = 0; ct < 12; ++ct) {
    v8f acc = zero8();
#pragma unroll
    for (int kk = 0; kk < 3; ++kk)
      acc = wmma_f16(ldA(sXs, CC, r0, kk * 32, lane),
                     ldB_rowk(kvw, CC, ct * 16, kk * 32, lane), acc);
    int n = ct * 16 + nlo; float bb = kvb[n]; int rb = r0 + hi8;
    if (ct < 6) {
#pragma unroll
      for (int r = 0; r < 8; ++r) sK[(rb + r) * CC + n] = (_Float16)(acc[r] + bb);
    } else {
      int vc = n - CC;
#pragma unroll
      for (int r = 0; r < 8; ++r) sVt[vc * TOK + rb + r] = (_Float16)(acc[r] + bb);
    }
  }
  __syncthreads();

  const float* mbase = mask + (size_t)(widx & 511) * (TOK * TOK);

  for (int h = 0; h < NHEAD; ++h) {
    // S = (Q_h*scale) @ K_h^T  (K padded 16->32 with zeros)
    for (int ct = 0; ct < 4; ++ct) {
      v8f acc = zero8();
      acc = wmma_f16(ldA_headQ(sQ, r0, h, lane), ldB_headK(sK, ct * 16, h, lane), acc);
      int n = ct * 16 + nlo; int rb = r0 + hi8;
#pragma unroll
      for (int r = 0; r < 8; ++r) sS[(rb + r) * TOK + n] = acc[r];
    }
    __syncthreads();

    // softmax rows (one thread per query row)
    if (tid < TOK) {
      const float* mrow = mbase + tid * TOK;
      float mx = -1e30f;
      for (int k = 0; k < TOK; ++k) mx = fmaxf(mx, sS[tid * TOK + k] + mrow[k]);
      float sum = 0.f;
      for (int k = 0; k < TOK; ++k) {
        float e = __expf(sS[tid * TOK + k] + mrow[k] - mx);
        sum += e;
        sP[tid * TOK + k] = (_Float16)e;
      }
      float inv = 1.f / sum;
      for (int k = 0; k < TOK; ++k)
        sP[tid * TOK + k] = (_Float16)((float)sP[tid * TOK + k] * inv);
    }
    __syncthreads();

    // O_h = P @ V_h  (B from transposed V: contiguous K per lane)
    {
      v8f acc = zero8();
#pragma unroll
      for (int kk = 0; kk < 2; ++kk)
        acc = wmma_f16(ldA(sP, TOK, r0, kk * 32, lane),
                       ldB_rowk(sVt, TOK, h * HD, kk * 32, lane), acc);
      int n = nlo; int rb = r0 + hi8;
#pragma unroll
      for (int r = 0; r < 8; ++r) sO[(rb + r) * CC + h * HD + n] = (_Float16)acc[r];
    }
    __syncthreads();
  }

  // Y = O @ proj^T + pb
  for (int ct = 0; ct < 6; ++ct) {
    v8f acc = zero8();
#pragma unroll
    for (int kk = 0; kk < 3; ++kk)
      acc = wmma_f16(ldA(sO, CC, r0, kk * 32, lane),
                     ldB_rowk(pw, CC, ct * 16, kk * 32, lane), acc);
    int n = ct * 16 + nlo; float bb = pb[n]; int rb = r0 + hi8;
#pragma unroll
    for (int r = 0; r < 8; ++r) sY[(rb + r) * CC + n] = (_Float16)(acc[r] + bb);
  }
  __syncthreads();

  // F = [Xw | Y] @ fuse^T + fb  (K = 192)
  for (int ct = 0; ct < 6; ++ct) {
    v8f acc = zero8();
#pragma unroll
    for (int kk = 0; kk < 6; ++kk) {
      int k0 = kk * 32;
      v16h a = (k0 < CC) ? ldA(sXw, CC, r0, k0, lane) : ldA(sY, CC, r0, k0 - CC, lane);
      acc = wmma_f16(a, ldB_rowk(fw, 192, ct * 16, k0, lane), acc);
    }
    int n = ct * 16 + nlo; float bb = fb[n]; int rb = r0 + hi8;
#pragma unroll
    for (int r = 0; r < 8; ++r) gF[(size_t)(rb + r) * CC + n] = acc[r] + bb;
  }
}

// ---------------- kernel 5: window reverse + residual + LN2 + WMMA MLP ----------------
#define TM_X1 0                       // float [64][96]
#define TM_LN 24576                   // f16   [64][96]
#define TM_H  36864                   // f16   [64][384]
#define SM_TAIL 86016

__global__ __launch_bounds__(128)
void ck_tail(const float* __restrict__ xl, const float* __restrict__ xr,
             const float* __restrict__ Fl, const float* __restrict__ Fr,
             const float* __restrict__ g2, const float* __restrict__ b2,
             const _Float16* __restrict__ fc1w, const float* __restrict__ fc1b,
             const _Float16* __restrict__ fc2w, const float* __restrict__ fc2b,
             float* __restrict__ out) {
  extern __shared__ char smem[];
  float*    sX1 = (float*)   (smem + TM_X1);
  _Float16* sLN = (_Float16*)(smem + TM_LN);
  _Float16* sH  = (_Float16*)(smem + TM_H);

  int dir = blockIdx.x >> 11;
  int blk = blockIdx.x & 2047;                 // 64-token tile over (b, n)
  const float* X = dir ? xr : xl;
  const float* F = dir ? Fr : Fl;
  float* O = out + (size_t)dir * NB * HH * WWI * CC;

  int tid = threadIdx.x, wave = tid >> 5, lane = tid & 31;
  int r0 = wave * 16, nlo = lane & 15, hi8 = (lane >> 4) << 3;

  // gather fuse output (inverse roll + window reverse), residual, LN2
  if (tid < TOK) {
    int tt = blk * TOK + tid;
    int b = tt >> 15, n = tt & 32767;
    int y = n >> 8, x = n & 255;
    int yw = (y + 124) & 127, xw = (x + 252) & 255;   // roll back by +ss
    int widx = b * NWPI + (yw >> 3) * 32 + (xw >> 3);
    int tw = (yw & 7) * 8 + (xw & 7);
    const float* frow = F + ((size_t)widx * TOK + tw) * CC;
    const float* xrow = X + (size_t)tt * CC;
    float s = 0.f, s2 = 0.f;
    for (int c = 0; c < CC; ++c) {
      float v = xrow[c] + frow[c];
      sX1[tid * CC + c] = v;
      s += v; s2 += v * v;
    }
    float m = s * (1.f / 96.f);
    float var = s2 * (1.f / 96.f) - m * m;
    float inv = rsqrtf(var + 1e-5f);
    for (int c = 0; c < CC; ++c)
      sLN[tid * CC + c] = (_Float16)((sX1[tid * CC + c] - m) * inv * g2[c] + b2[c]);
  }
  __syncthreads();

  // H = gelu(LN @ fc1^T + fc1b)
  for (int ct = 0; ct < 24; ++ct) {
    v8f acc = zero8();
#pragma unroll
    for (int kk = 0; kk < 3; ++kk)
      acc = wmma_f16(ldA(sLN, CC, r0, kk * 32, lane),
                     ldB_rowk(fc1w, CC, ct * 16, kk * 32, lane), acc);
    int n = ct * 16 + nlo; float bb = fc1b[n]; int rb = r0 + hi8;
#pragma unroll
    for (int r = 0; r < 8; ++r) {
      float v = acc[r] + bb;
      float ge = 0.5f * v * (1.f + erff(v * 0.70710678f));   // exact gelu
      sH[(rb + r) * HID + n] = (_Float16)ge;
    }
  }
  __syncthreads();

  // out = x1 + H @ fc2^T + fc2b
  for (int ct = 0; ct < 6; ++ct) {
    v8f acc = zero8();
#pragma unroll
    for (int kk = 0; kk < 12; ++kk)
      acc = wmma_f16(ldA(sH, HID, r0, kk * 32, lane),
                     ldB_rowk(fc2w, HID, ct * 16, kk * 32, lane), acc);
    int n = ct * 16 + nlo; float bb = fc2b[n]; int rb = r0 + hi8;
#pragma unroll
    for (int r = 0; r < 8; ++r) {
      int lr = rb + r;
      size_t tt = (size_t)blk * TOK + lr;
      O[tt * CC + n] = sX1[lr * CC + n] + acc[r] + bb;
    }
  }
}

// ---------------- host launcher ----------------
extern "C" void kernel_launch(void* const* d_in, const int* in_sizes, int n_in,
                              void* d_out, int out_size, void* d_ws, size_t ws_size,
                              hipStream_t stream) {
  const float* x_left  = (const float*)d_in[0];
  const float* x_right = (const float*)d_in[1];
  const int*   d_left  = (const int*)d_in[2];
  const int*   d_right = (const int*)d_in[3];
  const float* norm1_g = (const float*)d_in[6];
  const float* norm1_b = (const float*)d_in[7];
  const float* q_w  = (const float*)d_in[8];
  const float* q_b  = (const float*)d_in[9];
  const float* kv_w = (const float*)d_in[10];
  const float* kv_b = (const float*)d_in[11];
  const float* proj_w = (const float*)d_in[12];
  const float* proj_b = (const float*)d_in[13];
  const float* fuse_w = (const float*)d_in[14];
  const float* fuse_b = (const float*)d_in[15];
  const float* norm2_g = (const float*)d_in[16];
  const float* norm2_b = (const float*)d_in[17];
  const float* fc1_w = (const float*)d_in[18];
  const float* fc1_b = (const float*)d_in[19];
  const float* fc2_w = (const float*)d_in[20];
  const float* fc2_b = (const float*)d_in[21];

  char* w = (char*)d_ws;
  size_t off = 0;
  auto take = [&](size_t bytes) -> char* {
    char* p = w + off;
    off += (bytes + 255) & ~(size_t)255;
    return p;
  };
  float*    mask  = (float*)take((size_t)NWPI * TOK * TOK * 4);
  _Float16* qw16  = (_Float16*)take(9216 * 2);
  _Float16* kvw16 = (_Float16*)take(18432 * 2);
  _Float16* pw16  = (_Float16*)take(9216 * 2);
  _Float16* fw16  = (_Float16*)take(18432 * 2);
  _Float16* f1w16 = (_Float16*)take(36864 * 2);
  _Float16* f2w16 = (_Float16*)take(36864 * 2);
  size_t winBytes = (size_t)NWIN * TOK * CC * 2;
  _Float16* Xlw  = (_Float16*)take(winBytes);
  _Float16* Xrw  = (_Float16*)take(winBytes);
  _Float16* Xlsw = (_Float16*)take(winBytes);
  _Float16* Xrsw = (_Float16*)take(winBytes);
  float* Fl = (float*)take((size_t)NWIN * TOK * CC * 4);
  float* Fr = (float*)take((size_t)NWIN * TOK * CC * 4);

  ck_mask<<<(NWPI * TOK * TOK + 255) / 256, 256, 0, stream>>>(mask);
  ck_f2h<<<(9216  + 255) / 256, 256, 0, stream>>>(q_w,    qw16,  9216);
  ck_f2h<<<(18432 + 255) / 256, 256, 0, stream>>>(kv_w,   kvw16, 18432);
  ck_f2h<<<(9216  + 255) / 256, 256, 0, stream>>>(proj_w, pw16,  9216);
  ck_f2h<<<(18432 + 255) / 256, 256, 0, stream>>>(fuse_w, fw16,  18432);
  ck_f2h<<<(36864 + 255) / 256, 256, 0, stream>>>(fc1_w,  f1w16, 36864);
  ck_f2h<<<(36864 + 255) / 256, 256, 0, stream>>>(fc2_w,  f2w16, 36864);

  ck_prep<<<65536, 128, 0, stream>>>(x_left, x_right, d_left, d_right,
                                     norm1_g, norm1_b, Xlw, Xrw, Xlsw, Xrsw);

  (void)hipFuncSetAttribute(reinterpret_cast<const void*>(ck_attn),
                            hipFuncAttributeMaxDynamicSharedMemorySize, SM_ATTN);
  ck_attn<<<2 * NWIN, 128, SM_ATTN, stream>>>(Xlw, Xrw, Xlsw, Xrsw, mask,
                                              qw16, q_b, kvw16, kv_b,
                                              pw16, proj_b, fw16, fuse_b, Fl, Fr);

  (void)hipFuncSetAttribute(reinterpret_cast<const void*>(ck_tail),
                            hipFuncAttributeMaxDynamicSharedMemorySize, SM_TAIL);
  ck_tail<<<2 * NWIN, 128, SM_TAIL, stream>>>(x_left, x_right, Fl, Fr,
                                              norm2_g, norm2_b,
                                              f1w16, fc1_b, f2w16, fc2_b,
                                              (float*)d_out);
}
